// GE2ELoss_429496730178
// MI455X (gfx1250) — matrix operand order repaired and checked by reading, not compiled, not run on hardware
//
#include <hip/hip_runtime.h>

// GE2E softmax loss, fused:
//   K0: zero loss accumulator
//   K1 (per speaker n): column sums -> normalized bf16 centroid; per-utt
//       |x|^2, dot(x,sum) -> closed-form leave-one-out cosine (cos_own)
//   K2 (per 16-row strip): row-normalize x to bf16 in LDS, BF16 WMMA GEMM
//       vs all 1024 centroids, fused clamp/affine/exp/logsumexp epilogue,
//       atomicAdd of per-strip loss sum
//   K3: scale by 1/(N*M) into d_out

typedef __attribute__((ext_vector_type(16))) __bf16 v16bf;
typedef __attribute__((ext_vector_type(8)))  __bf16 v8bf;
typedef __attribute__((ext_vector_type(8)))  float  v8f;

#define GN 1024
#define GM 32
#define GD 512
#define STRIDE_A 520   // padded LDS row stride (bf16 elems); 1040 B = 65*16 B

// ---------------------------------------------------------------- K0 / K3
__global__ void ge2e_zero(float* acc) {
    if (threadIdx.x == 0) *acc = 0.0f;
}
__global__ void ge2e_fin(const float* __restrict__ acc, float* __restrict__ out) {
    *out = *acc * (1.0f / (float)(GN * GM));
}

// ---------------------------------------------------------------- K1: prep
__global__ __launch_bounds__(256)
void ge2e_prep(const float* __restrict__ x,
               __bf16* __restrict__ cbf,      // (GN, GD) unit-norm centroids
               float* __restrict__ cosOwn)    // (GN, GM) leave-one-out cosine
{
    __shared__ float ssum[GD];
    __shared__ float red[8];
    __shared__ float snorm2s;

    const int n    = blockIdx.x;
    const int tid  = threadIdx.x;
    const int lane = tid & 31;
    const int wave = tid >> 5;
    const float* xn = x + (size_t)n * GM * GD;

    // column sums over the M utterances (coalesced across threads)
    const int d0 = tid, d1 = tid + 256;
    float a0 = 0.0f, a1 = 0.0f;
    for (int m = 0; m < GM; ++m) {
        a0 += xn[m * GD + d0];
        a1 += xn[m * GD + d1];
    }
    ssum[d0] = a0; ssum[d1] = a1;

    // |sum|^2: block reduction
    float p = a0 * a0 + a1 * a1;
    p += __shfl_xor(p, 1, 32);  p += __shfl_xor(p, 2, 32);
    p += __shfl_xor(p, 4, 32);  p += __shfl_xor(p, 8, 32);
    p += __shfl_xor(p, 16, 32);
    if (lane == 0) red[wave] = p;
    __syncthreads();
    if (tid == 0) {
        float s = 0.0f;
        for (int i = 0; i < 8; ++i) s += red[i];
        snorm2s = s;
    }
    __syncthreads();
    const float snorm2 = snorm2s;

    // per-utterance |x|^2 and dot(x, sum): 8 threads per m
    const int m = tid >> 3, g = tid & 7;
    float a2 = 0.0f, ds = 0.0f;
    for (int d = g; d < GD; d += 8) {
        const float xv = xn[m * GD + d];
        a2 += xv * xv;
        ds += xv * ssum[d];
    }
    a2 += __shfl_xor(a2, 1, 32); a2 += __shfl_xor(a2, 2, 32); a2 += __shfl_xor(a2, 4, 32);
    ds += __shfl_xor(ds, 1, 32); ds += __shfl_xor(ds, 2, 32); ds += __shfl_xor(ds, 4, 32);
    if (g == 0) {
        // cos(x, (sum-x)/(M-1)): (M-1) factors cancel
        const float num  = ds - a2;
        const float den2 = snorm2 - 2.0f * ds + a2;
        cosOwn[n * GM + m] = num * rsqrtf(fmaxf(a2, 1e-30f))
                                 * rsqrtf(fmaxf(den2, 1e-30f));
    }

    // unit-norm centroid in bf16: sum / |sum|
    const float rs = rsqrtf(fmaxf(snorm2, 1e-30f));
    cbf[(size_t)n * GD + d0] = (__bf16)(ssum[d0] * rs);
    cbf[(size_t)n * GD + d1] = (__bf16)(ssum[d1] * rs);
}

// ------------------------------------------- K2: WMMA GEMM + fused softmax
__global__ __launch_bounds__(256)
void ge2e_gemm_softmax(const float* __restrict__ x,
                       const __bf16* __restrict__ cbf,
                       const float* __restrict__ cosOwn,
                       const float* __restrict__ wp,
                       const float* __restrict__ bp,
                       float* __restrict__ acc)
{
    __shared__ __align__(32) __bf16 Asmem[16 * STRIDE_A];
    __shared__ float rowrs[16];
    __shared__ float rowsum[16];
    __shared__ float csown[16];

    const int strip = blockIdx.x;          // 2048 strips of 16 rows
    const int n     = strip >> 1;          // speaker (16 rows share n)
    const int mbase = (strip & 1) << 4;
    const int tid   = threadIdx.x;
    const int wave  = tid >> 5;
    const int lane  = tid & 31;
    const int hi    = lane >> 4;           // half-wave select
    const int lr    = lane & 15;           // A row / B col for this lane

    const float* xs = x + (size_t)(n * GM + mbase) * GD;

    // pass 1: per-row inverse norms (16 threads per row)
    {
        const int r = tid >> 4, g = tid & 15;
        float s = 0.0f;
        for (int d = g; d < GD; d += 16) { const float v = xs[r * GD + d]; s += v * v; }
        s += __shfl_xor(s, 1, 32); s += __shfl_xor(s, 2, 32);
        s += __shfl_xor(s, 4, 32); s += __shfl_xor(s, 8, 32);
        if (g == 0) rowrs[r] = rsqrtf(fmaxf(s, 1e-30f));
    }
    if (tid < 16) { rowsum[tid] = 0.0f; csown[tid] = 0.0f; }
    __syncthreads();

    // pass 2: stage unit-norm bf16 A strip into LDS (padded stride)
    for (int idx = tid; idx < 16 * GD; idx += 256) {
        const int r = idx >> 9, d = idx & (GD - 1);
        Asmem[r * STRIDE_A + d] = (__bf16)(xs[r * GD + d] * rowrs[r]);
    }
    __syncthreads();

    const float w = *wp, b = *bp;
    const __bf16* abase = &Asmem[lr * STRIDE_A + (hi << 4)];

    float se[8];
    #pragma unroll
    for (int v = 0; v < 8; ++v) se[v] = 0.0f;

    // each wave owns 8 of the 64 column tiles
    for (int j = 0; j < 8; ++j) {
        const int ct = wave + (j << 3);
        const __bf16* bbase = cbf + (size_t)(ct * 16 + lr) * GD + (hi << 4);
        v8f accv = {};
        #pragma unroll
        for (int ks = 0; ks < 16; ++ks) {
            // A frag 16x32 bf16: lane=row, two 16B chunks (contiguous K = 16*hi + e)
            const v8bf alo = *(const v8bf*)(abase + ks * 32);
            const v8bf ahi = *(const v8bf*)(abase + ks * 32 + 8);
            v16bf afrag;
            #pragma unroll
            for (int e = 0; e < 8; ++e) { afrag[e] = alo[e]; afrag[e + 8] = ahi[e]; }
            // B frag 32x16 bf16: lane=col, 16 contiguous bf16 (32B)
            const v16bf bfrag = *(const v16bf*)(bbase + ks * 32);
            accv = __builtin_amdgcn_wmma_f32_16x16x32_bf16(
                false, afrag, false, bfrag, (short)0, accv, false, false);
        }
        // fused epilogue: own-column swap, clamp, affine, exp-accumulate
        const int  gcol = (ct << 4) + lr;
        const bool own  = (gcol == n);
        #pragma unroll
        for (int v = 0; v < 8; ++v) {
            const int r = v + (hi << 3);                     // C layout row
            float val = own ? cosOwn[n * GM + mbase + r] : accv[v];
            val = fmaxf(val, 1e-6f);
            const float cs = __builtin_fmaf(w, val, b);      // cs in [-5, 5]
            if (own) csown[r] = cs;                          // unique writer
            se[v] += __expf(cs);                             // no overflow
        }
    }

    // row sum-of-exp: reduce 16 lanes of each half-wave, combine via LDS
    #pragma unroll
    for (int v = 0; v < 8; ++v) {
        float s = se[v];
        s += __shfl_xor(s, 1, 32); s += __shfl_xor(s, 2, 32);
        s += __shfl_xor(s, 4, 32); s += __shfl_xor(s, 8, 32);
        if (lr == 0) atomicAdd(&rowsum[v + (hi << 3)], s);
    }
    __syncthreads();

    // L[r] = log(sum_exp) - cs_own ; sum over the 16 rows, one global atomic
    if (tid < 16) {
        float L = __logf(rowsum[tid]) - csown[tid];
        L += __shfl_xor(L, 1, 32); L += __shfl_xor(L, 2, 32);
        L += __shfl_xor(L, 4, 32); L += __shfl_xor(L, 8, 32);
        if (tid == 0) atomicAdd(acc, L);
    }
}

// ---------------------------------------------------------------- launch
extern "C" void kernel_launch(void* const* d_in, const int* in_sizes, int n_in,
                              void* d_out, int out_size, void* d_ws, size_t ws_size,
                              hipStream_t stream)
{
    const float* x = (const float*)d_in[0];
    const float* w = (const float*)d_in[1];
    const float* b = (const float*)d_in[2];

    char* ws = (char*)d_ws;
    float*  acc    = (float*)(ws);                               // 4 B
    float*  cosOwn = (float*)(ws + 1024);                        // 128 KiB
    __bf16* cbf    = (__bf16*)(ws + 1024 + (size_t)GN * GM * 4); // 1 MiB, 32B-aligned

    ge2e_zero<<<1, 32, 0, stream>>>(acc);
    ge2e_prep<<<GN, 256, 0, stream>>>(x, cbf, cosOwn);
    ge2e_gemm_softmax<<<(GN * GM) / 16, 256, 0, stream>>>(x, cbf, cosOwn, w, b, acc);
    ge2e_fin<<<1, 1, 0, stream>>>(acc, (float*)d_out);
}
